// ExtendedTransformer1_5274219840241
// MI455X (gfx1250) — matrix operand order repaired and checked by reading, not compiled, not run on hardware
//
// ExtendedTransformer forward for MI455X (gfx1250), compile-only target.
//
// Strategy: the network is matrix-core bound (~280 GFLOP of GEMMs vs ~300MB
// of unique bytes -> ~15us of HBM traffic at 23.3 TB/s, vs ~100us of WMMA
// math). Every matmul goes through v_wmma_f32_16x16x32_f16 with fp32
// accumulation; fp32->f16 conversion happens while staging tiles into LDS
// using packed v_cvt_pk converts. The GEMM is register-staged double-
// buffered: float4 (b128) global loads of tile k+1 are issued while tile k's
// WMMAs run. transB (QK^T) is a separate template instantiation so the
// common GEMM hot loop carries no scatter branch. Softmax / LayerNorm /
// relative-bias terms are cheap VALU kernels.
//
// Input flattening assumption (JAX pytree = sorted dict keys):
//  0:src 1:trg 2:enc_rel 3:dec_rel 4:d2e_rel
//  dec.cattn{Bb,E,bk,bo,bq,bv,wk,wo,wq,wv}=5..14  dec.ffn{b1,b2,w1,w2}=15..18
//  dec.ln1{b,g}=19,20 ln2=21,22 ln3=23,24  dec.sattn=25..34
//  35:emb_dec 36:emb_enc  enc.attn=37..46 enc.ffn=47..50 enc.ln1=51,52
//  enc.ln2=53,54  55:fc_b 56:fc_w

#include <hip/hip_runtime.h>
#include <hip/hip_bf16.h>

#define B_    8
#define SRC_  512
#define TRG_  256
#define D_    512
#define H_    8
#define L_    4
#define F_    2048
#define V_    32000
#define NREL_ 33
#define DK_   64

typedef __attribute__((ext_vector_type(16))) _Float16 v16h;
typedef __attribute__((ext_vector_type(2)))  __fp16   v2fp;   // cvt_pkrtz result type
typedef __attribute__((ext_vector_type(8)))  float    v8f;

union Frag16 { v16h h; uint4 q[2]; };
union PK2    { v2fp h; unsigned u; _Float16 s[2]; };

static constexpr int BM = 64;
static constexpr int BN = 128;
static constexpr int BK = 32;

// Packed fp32x2 -> f16x2 convert (v_cvt_pk_rtz_f16_f32): 1 VALU per pair.
__device__ inline uint2 cvt4_f16(float4 f) {
    PK2 lo, hi;
    lo.h = __builtin_amdgcn_cvt_pkrtz(f.x, f.y);
    hi.h = __builtin_amdgcn_cvt_pkrtz(f.z, f.w);
    uint2 r; r.x = lo.u; r.y = hi.u;
    return r;
}

// ---------------------------------------------------------------------------
// Generic batched GEMM:  C[z] = act( A[z] (.) B[z](^T) + bias )
// A fp32 [M,K] (row stride lda), B fp32 [K,N] (or [N,K] if TRANSB) , C fp32.
// Per-z offsets decompose z = zb*Hdiv + zh with independent (b,h) strides so
// one kernel covers plain GEMMs, per-(b,h) QK^T and per-(b,h) A*V.
// 256 threads = 8 waves; each wave computes a 32x32 patch = 2x2 WMMA tiles.
// Register-staged double buffering: tile k+1 loaded as float4 while tile k
// computes; packed fp32->f16 conversion at the LDS store.
// ---------------------------------------------------------------------------
template <int TRANSB>
__global__ __launch_bounds__(256)
void gemm_f16_kernel(const float* __restrict__ A, const float* __restrict__ Bm,
                     const float* __restrict__ bias, float* __restrict__ C,
                     int M, int N, int K, int lda, int ldb, int ldc,
                     long long sA1, long long sA2, long long sB1, long long sB2,
                     long long sC1, long long sC2, int Hdiv, int relu)
{
    // Padded so fragment rows stay 16B aligned: 40*2=80B, 136*2=272B strides.
    __shared__ __align__(16) _Float16 As[BM][BK + 8];
    __shared__ __align__(16) _Float16 Bs[BK][BN + 8];

    const int tid   = threadIdx.x;
    const int lane  = tid & 31;
    const int wave  = tid >> 5;
    const int waveM = wave >> 2;     // 0..1  (rows of 32)
    const int waveN = wave & 3;      // 0..3  (cols of 32)
    const int grp   = lane >> 4;     // A k-half / C m-group
    const int mrow  = lane & 15;

    const int m0 = blockIdx.y * BM;
    const int n0 = blockIdx.x * BN;
    const int z  = blockIdx.z;
    const int zb = z / Hdiv, zh = z % Hdiv;
    const long long offA = (long long)zb * sA1 + (long long)zh * sA2;
    const long long offB = (long long)zb * sB1 + (long long)zh * sB2;
    const long long offC = (long long)zb * sC1 + (long long)zh * sC2;

    const float* Ab  = A + offA;
    const float* Bb2 = Bm + offB;

    // Per-thread staging slots (register double buffer):
    //  A tile 64x32 fp32 = 512 float4 -> 2 per thread.
    //  B tile 32x128 fp32 = 1024 float4 -> 4 per thread.
    float4 aReg[2], bReg[4];

    // A quad j: flat quad id qa = tid + j*256 -> row r = qa>>3, col4 = (qa&7)*4
    // B quad j (TRANSB=0): qb = tid + j*256 -> row k = qb>>5, col4 = (qb&31)*4
    // B quad j (TRANSB=1): qb -> n = qb>>3, k4 = (qb&7)*4  (contiguous in k!)
    auto load_tiles = [&](int k0) {
#pragma unroll
        for (int j = 0; j < 2; ++j) {
            int qa = tid + j * 256;
            int r  = qa >> 3, c4 = (qa & 7) * 4;
            int gm = m0 + r, gk = k0 + c4;
            float4 v = make_float4(0.f, 0.f, 0.f, 0.f);
            if (gm < M && gk + 3 < K) {
                v = *(const float4*)(Ab + (long long)gm * lda + gk);
            } else if (gm < M) {
                const float* ap = Ab + (long long)gm * lda;
                if (gk + 0 < K) v.x = ap[gk + 0];
                if (gk + 1 < K) v.y = ap[gk + 1];
                if (gk + 2 < K) v.z = ap[gk + 2];
                if (gk + 3 < K) v.w = ap[gk + 3];
            }
            aReg[j] = v;
        }
#pragma unroll
        for (int j = 0; j < 4; ++j) {
            int qb = tid + j * 256;
            float4 v = make_float4(0.f, 0.f, 0.f, 0.f);
            if (!TRANSB) {
                int r = qb >> 5, c4 = (qb & 31) * 4;
                int gk = k0 + r, gn = n0 + c4;
                if (gk < K) {
                    const float* bp = Bb2 + (long long)gk * ldb;
                    if (gn + 3 < N) {
                        v = *(const float4*)(bp + gn);
                    } else {
                        if (gn + 0 < N) v.x = bp[gn + 0];
                        if (gn + 1 < N) v.y = bp[gn + 1];
                        if (gn + 2 < N) v.z = bp[gn + 2];
                        if (gn + 3 < N) v.w = bp[gn + 3];
                    }
                }
            } else {
                int nn = qb >> 3, k4 = (qb & 7) * 4;
                int gn = n0 + nn, gk = k0 + k4;
                if (gn < N && gk + 3 < K)
                    v = *(const float4*)(Bb2 + (long long)gn * ldb + gk);
            }
            bReg[j] = v;
        }
    };

    auto store_tiles = [&]() {
#pragma unroll
        for (int j = 0; j < 2; ++j) {
            int qa = tid + j * 256;
            int r  = qa >> 3, c4 = (qa & 7) * 4;
            *(uint2*)&As[r][c4] = cvt4_f16(aReg[j]);
        }
#pragma unroll
        for (int j = 0; j < 4; ++j) {
            int qb = tid + j * 256;
            if (!TRANSB) {
                int r = qb >> 5, c4 = (qb & 31) * 4;
                *(uint2*)&Bs[r][c4] = cvt4_f16(bReg[j]);
            } else {
                int nn = qb >> 3, k4 = (qb & 7) * 4;
                PK2 lo, hi;
                lo.h = __builtin_amdgcn_cvt_pkrtz(bReg[j].x, bReg[j].y);
                hi.h = __builtin_amdgcn_cvt_pkrtz(bReg[j].z, bReg[j].w);
                Bs[k4 + 0][nn] = lo.s[0];
                Bs[k4 + 1][nn] = lo.s[1];
                Bs[k4 + 2][nn] = hi.s[0];
                Bs[k4 + 3][nn] = hi.s[1];
            }
        }
    };

    v8f acc[2][2] = {};
    load_tiles(0);

    for (int k0 = 0; k0 < K; k0 += BK) {
        __syncthreads();          // previous iteration's fragment reads done
        store_tiles();
        __syncthreads();

        // Issue next tile's global loads now; their waits land at the next
        // store_tiles(), overlapping with the WMMAs below.
        if (k0 + BK < K) load_tiles(k0 + BK);
        // And prefetch two tiles ahead (gfx1250 global_prefetch_b8 path).
        if (k0 + 2 * BK < K) {
            int gm = m0 + (tid & (BM - 1));
            if (gm < M) __builtin_prefetch(Ab + (long long)gm * lda + k0 + 2 * BK, 0, 1);
        }

        // Fragments per ISA 7.12.2 layouts.
        // A (16x32 f16): lane(m)=mrow, lane-half selects k-block {0..7}/{8..15};
        // elements 0..7 -> k = grp*8..+7, elements 8..15 -> k = 16+grp*8..+7.
        Frag16 af[2], bf[2];
#pragma unroll
        for (int ma = 0; ma < 2; ++ma) {
            int m = waveM * 32 + ma * 16 + mrow;
            af[ma].q[0] = *(const uint4*)&As[m][grp * 8];
            af[ma].q[1] = *(const uint4*)&As[m][16 + grp * 8];
        }
        // B (32x16 f16): lane(n)=mrow; elements 0..7 -> k = grp*8..+7, 8..15 -> k = 16+grp*8..+7.
#pragma unroll
        for (int nb = 0; nb < 2; ++nb) {
            int nOff = waveN * 32 + nb * 16;
#pragma unroll
            for (int s = 0; s < 8; ++s) {
                bf[nb].h[s]     = Bs[grp * 8 + s][nOff + mrow];
                bf[nb].h[s + 8] = Bs[16 + grp * 8 + s][nOff + mrow];
            }
        }
#pragma unroll
        for (int ma = 0; ma < 2; ++ma)
#pragma unroll
            for (int nb = 0; nb < 2; ++nb)
                acc[ma][nb] = __builtin_amdgcn_wmma_f32_16x16x32_f16(
                    false, af[ma].h, false, bf[nb].h,
                    (short)0, acc[ma][nb], false, false);
    }

    // Store: C/D layout -> lane = n (mod 16), lane-half picks m-group of 8,
    // vector element v = m within group.
#pragma unroll
    for (int ma = 0; ma < 2; ++ma)
#pragma unroll
        for (int nb = 0; nb < 2; ++nb) {
            int mloc = waveM * 32 + ma * 16 + grp * 8;
            int gn   = n0 + waveN * 32 + nb * 16 + (lane & 15);
            if (gn >= N) continue;
            float bv = bias ? bias[gn] : 0.f;
#pragma unroll
            for (int v = 0; v < 8; ++v) {
                int gm = m0 + mloc + v;
                if (gm >= M) continue;
                float val = acc[ma][nb][v] + bv;
                if (relu) val = fmaxf(val, 0.f);
                C[offC + (long long)gm * ldc + gn] = val;
            }
        }
}

// ---------------------------------------------------------------------------
// Fused relative-position terms + masking + softmax, in-place on S.
// S[bh, q, k] <- softmax_k( (S + Q.E[rel]^T + Bb[h,rel]) * scale - 1e9*mask )
// One block (256 thr) per (q, bh) row. The 33 Q.E[j] dots go to LDS first.
// ---------------------------------------------------------------------------
__global__ __launch_bounds__(256)
void rel_softmax_kernel(float* __restrict__ S, const float* __restrict__ Q,
                        const float* __restrict__ E, const float* __restrict__ Bb,
                        const int* __restrict__ rel, const int* __restrict__ ktok,
                        int causal, int sq, int sk, float scale)
{
    const int q   = blockIdx.x;
    const int bh  = blockIdx.y;
    const int b   = bh >> 3;            // H_ == 8
    const int h   = bh & 7;
    const int tid = threadIdx.x;

    __shared__ float Rb[NREL_];
    __shared__ float red[256];

    float*       Srow = S + ((long long)bh * sq + q) * sk;
    const float* Qrow = Q + ((long long)(b * sq + q)) * D_ + h * DK_;

    if (tid < NREL_) {
        float r = 0.f;
        const float* Ej = E + tid * DK_;
#pragma unroll 8
        for (int d = 0; d < DK_; ++d) r += Qrow[d] * Ej[d];
        Rb[tid] = r + Bb[h * NREL_ + tid];
    }
    __syncthreads();

    float loc[2];
    int   kn = 0;
    float mx = -3.4e38f;
    for (int k = tid; k < sk; k += 256) {
        float v = (Srow[k] + Rb[rel[q * sk + k]]) * scale;
        float m = 0.f;
        if (causal && k > q) m = 1.f;
        if (ktok && ktok[b * sk + k] == 0) m = 1.f;   // padding mask
        v += m * -1e9f;
        loc[kn++] = v;
        mx = fmaxf(mx, v);
    }
    red[tid] = mx; __syncthreads();
    for (int o = 128; o > 0; o >>= 1) { if (tid < o) red[tid] = fmaxf(red[tid], red[tid + o]); __syncthreads(); }
    mx = red[0]; __syncthreads();

    float sum = 0.f;
    for (int i = 0; i < kn; ++i) { loc[i] = __expf(loc[i] - mx); sum += loc[i]; }
    red[tid] = sum; __syncthreads();
    for (int o = 128; o > 0; o >>= 1) { if (tid < o) red[tid] += red[tid + o]; __syncthreads(); }
    float inv = 1.f / red[0];

    kn = 0;
    for (int k = tid; k < sk; k += 256) Srow[k] = loc[kn++] * inv;
}

// ---------------------------------------------------------------------------
// Embedding gather: out[row,:] = emb[ids[row],:]
// ---------------------------------------------------------------------------
__global__ __launch_bounds__(128)
void embed_kernel(const int* __restrict__ ids, const float* __restrict__ emb,
                  float* __restrict__ out)
{
    const int row = blockIdx.x;
    const int id  = ids[row];
    for (int c = threadIdx.x * 4; c < D_; c += 128 * 4) {
        float4 v = *(const float4*)(emb + (long long)id * D_ + c);
        *(float4*)(out + (long long)row * D_ + c) = v;
    }
}

// ---------------------------------------------------------------------------
// Residual + LayerNorm (in place on x): x <- LN(x + h) * g + b
// One 128-thread block per token row (D=512 -> 4 elems/thread).
// ---------------------------------------------------------------------------
__global__ __launch_bounds__(128)
void add_ln_kernel(float* __restrict__ x, const float* __restrict__ h,
                   const float* __restrict__ g, const float* __restrict__ bta)
{
    const int row = blockIdx.x;
    const int tid = threadIdx.x;
    __shared__ float red[128];

    float v[4];
    float s = 0.f;
#pragma unroll
    for (int i = 0; i < 4; ++i) {
        int c = tid + i * 128;
        v[i]  = x[(long long)row * D_ + c] + h[(long long)row * D_ + c];
        s    += v[i];
    }
    red[tid] = s; __syncthreads();
    for (int o = 64; o > 0; o >>= 1) { if (tid < o) red[tid] += red[tid + o]; __syncthreads(); }
    const float mean = red[0] * (1.f / D_);
    __syncthreads();

    float s2 = 0.f;
#pragma unroll
    for (int i = 0; i < 4; ++i) { float d = v[i] - mean; s2 += d * d; }
    red[tid] = s2; __syncthreads();
    for (int o = 64; o > 0; o >>= 1) { if (tid < o) red[tid] += red[tid + o]; __syncthreads(); }
    const float inv = rsqrtf(red[0] * (1.f / D_) + 1e-5f);

#pragma unroll
    for (int i = 0; i < 4; ++i) {
        int c = tid + i * 128;
        x[(long long)row * D_ + c] = (v[i] - mean) * inv * g[c] + bta[c];
    }
}

// ---------------------------------------------------------------------------
// Host-side orchestration
// ---------------------------------------------------------------------------
static inline void launch_gemm(hipStream_t st, const float* A, const float* Bm,
                               const float* bias, float* C,
                               int M, int N, int K, int lda, int ldb, int ldc,
                               long long sA1, long long sA2, long long sB1, long long sB2,
                               long long sC1, long long sC2,
                               int batches, int Hdiv, int transB, int relu)
{
    dim3 g((N + BN - 1) / BN, (M + BM - 1) / BM, batches);
    if (transB)
        gemm_f16_kernel<1><<<g, 256, 0, st>>>(A, Bm, bias, C, M, N, K, lda, ldb, ldc,
                                              sA1, sA2, sB1, sB2, sC1, sC2, Hdiv, relu);
    else
        gemm_f16_kernel<0><<<g, 256, 0, st>>>(A, Bm, bias, C, M, N, K, lda, ldb, ldc,
                                              sA1, sA2, sB1, sB2, sC1, sC2, Hdiv, relu);
}

struct AttnW {
    const float *wq, *bq, *wk, *bk, *wv, *bv, *wo, *bo, *E, *Bb;
};

static void run_attention(hipStream_t st, const AttnW& w,
                          const float* Xq, const float* Xkv, int sq, int sk,
                          const int* rel, const int* ktok, int causal,
                          float* Qb, float* Kb, float* Vb, float* Sb,
                          float* T1, float* Obuf)
{
    const int Mq = B_ * sq, Mk = B_ * sk;
    launch_gemm(st, Xq,  w.wq, w.bq, Qb, Mq, D_, D_, D_, D_, D_, 0,0,0,0,0,0, 1,1,0,0);
    launch_gemm(st, Xkv, w.wk, w.bk, Kb, Mk, D_, D_, D_, D_, D_, 0,0,0,0,0,0, 1,1,0,0);
    launch_gemm(st, Xkv, w.wv, w.bv, Vb, Mk, D_, D_, D_, D_, D_, 0,0,0,0,0,0, 1,1,0,0);
    // S[b,h] = Q[b,h] K[b,h]^T   (per-head strided views, B operand transposed)
    launch_gemm(st, Qb, Kb, nullptr, Sb, sq, sk, DK_, D_, D_, sk,
                (long long)sq * D_, DK_, (long long)sk * D_, DK_,
                (long long)H_ * sq * sk, (long long)sq * sk,
                B_ * H_, H_, 1, 0);
    rel_softmax_kernel<<<dim3(sq, B_ * H_), 256, 0, st>>>(
        Sb, Qb, w.E, w.Bb, rel, ktok, causal, sq, sk, 0.125f);
    // T1[b, q, h*64+d] = sum_k A[b,h,q,k] V[b,h,k,d]  (head-concat layout)
    launch_gemm(st, Sb, Vb, nullptr, T1, sq, DK_, sk, sk, D_, D_,
                (long long)H_ * sq * sk, (long long)sq * sk,
                (long long)sk * D_, DK_,
                (long long)sq * D_, DK_,
                B_ * H_, H_, 0, 0);
    launch_gemm(st, T1, w.wo, w.bo, Obuf, Mq, D_, D_, D_, D_, D_, 0,0,0,0,0,0, 1,1,0,0);
}

static void run_ffn(hipStream_t st, const float* X, int rows,
                    const float* w1, const float* b1,
                    const float* w2, const float* b2, float* T0, float* T1)
{
    launch_gemm(st, X,  w1, b1, T0, rows, F_, D_, D_, F_, F_, 0,0,0,0,0,0, 1,1,0,1);
    launch_gemm(st, T0, w2, b2, T1, rows, D_, F_, F_, D_, D_, 0,0,0,0,0,0, 1,1,0,0);
}

// Flattened input indices (sorted-dict-key pytree order; see header comment).
enum {
    IN_SRC = 0, IN_TRG, IN_ENC_REL, IN_DEC_REL, IN_D2E_REL,
    DC_BB = 5, DC_E, DC_BK, DC_BO, DC_BQ, DC_BV, DC_WK, DC_WO, DC_WQ, DC_WV,
    DF_B1 = 15, DF_B2, DF_W1, DF_W2,
    DL1_B = 19, DL1_G, DL2_B = 21, DL2_G, DL3_B = 23, DL3_G,
    DS_BB = 25, DS_E, DS_BK, DS_BO, DS_BQ, DS_BV, DS_WK, DS_WO, DS_WQ, DS_WV,
    EMB_DEC = 35, EMB_ENC = 36,
    EA_BB = 37, EA_E, EA_BK, EA_BO, EA_BQ, EA_BV, EA_WK, EA_WO, EA_WQ, EA_WV,
    EF_B1 = 47, EF_B2, EF_W1, EF_W2,
    EL1_B = 51, EL1_G, EL2_B = 53, EL2_G,
    FC_B = 55, FC_W = 56
};

extern "C" void kernel_launch(void* const* d_in, const int* in_sizes, int n_in,
                              void* d_out, int out_size, void* d_ws, size_t ws_size,
                              hipStream_t stream)
{
    (void)in_sizes; (void)n_in; (void)out_size; (void)ws_size;
    auto FP = [&](int i) { return (const float*)d_in[i]; };
    const int* src     = (const int*)d_in[IN_SRC];
    const int* trg     = (const int*)d_in[IN_TRG];
    const int* enc_rel = (const int*)d_in[IN_ENC_REL];
    const int* dec_rel = (const int*)d_in[IN_DEC_REL];
    const int* d2e_rel = (const int*)d_in[IN_D2E_REL];

    // Workspace carve-up (~147 MB of fp32 scratch).
    float* p = (float*)d_ws;
    auto take = [&](size_t n) { float* r = p; p += n; return r; };
    float* XE = take((size_t)B_ * SRC_ * D_);            // encoder activations
    float* XD = take((size_t)B_ * TRG_ * D_);            // decoder activations
    float* Qb = take((size_t)B_ * SRC_ * D_);
    float* Kb = take((size_t)B_ * SRC_ * D_);
    float* Vb = take((size_t)B_ * SRC_ * D_);
    float* T0 = take((size_t)B_ * SRC_ * F_);            // FFN hidden
    float* T1 = take((size_t)B_ * SRC_ * D_);
    float* Sb = take((size_t)B_ * H_ * SRC_ * SRC_);     // scores / probs

    const long long DD = (long long)D_ * D_;
    const long long DF = (long long)D_ * F_;

    // ---------------- Encoder ----------------
    embed_kernel<<<B_ * SRC_, 128, 0, stream>>>(src, FP(EMB_ENC), XE);
    for (int i = 0; i < L_; ++i) {
        AttnW w{ FP(EA_WQ) + i * DD, FP(EA_BQ) + i * D_,
                 FP(EA_WK) + i * DD, FP(EA_BK) + i * D_,
                 FP(EA_WV) + i * DD, FP(EA_BV) + i * D_,
                 FP(EA_WO) + i * DD, FP(EA_BO) + i * D_,
                 FP(EA_E)  + i * NREL_ * DK_, FP(EA_BB) + i * H_ * NREL_ };
        run_attention(stream, w, XE, XE, SRC_, SRC_, enc_rel, src, 0,
                      Qb, Kb, Vb, Sb, T1, Qb);
        add_ln_kernel<<<B_ * SRC_, 128, 0, stream>>>(XE, Qb, FP(EL1_G) + i * D_, FP(EL1_B) + i * D_);
        run_ffn(stream, XE, B_ * SRC_, FP(EF_W1) + i * DF, FP(EF_B1) + i * F_,
                FP(EF_W2) + i * DF, FP(EF_B2) + i * D_, T0, T1);
        add_ln_kernel<<<B_ * SRC_, 128, 0, stream>>>(XE, T1, FP(EL2_G) + i * D_, FP(EL2_B) + i * D_);
    }

    // ---------------- Decoder ----------------
    embed_kernel<<<B_ * TRG_, 128, 0, stream>>>(trg, FP(EMB_DEC), XD);
    for (int i = 0; i < L_; ++i) {
        AttnW ws_{ FP(DS_WQ) + i * DD, FP(DS_BQ) + i * D_,
                   FP(DS_WK) + i * DD, FP(DS_BK) + i * D_,
                   FP(DS_WV) + i * DD, FP(DS_BV) + i * D_,
                   FP(DS_WO) + i * DD, FP(DS_BO) + i * D_,
                   FP(DS_E)  + i * NREL_ * DK_, FP(DS_BB) + i * H_ * NREL_ };
        run_attention(stream, ws_, XD, XD, TRG_, TRG_, dec_rel, trg, 1,
                      Qb, Kb, Vb, Sb, T1, Qb);
        add_ln_kernel<<<B_ * TRG_, 128, 0, stream>>>(XD, Qb, FP(DL1_G) + i * D_, FP(DL1_B) + i * D_);

        AttnW wc{ FP(DC_WQ) + i * DD, FP(DC_BQ) + i * D_,
                  FP(DC_WK) + i * DD, FP(DC_BK) + i * D_,
                  FP(DC_WV) + i * DD, FP(DC_BV) + i * D_,
                  FP(DC_WO) + i * DD, FP(DC_BO) + i * D_,
                  FP(DC_E)  + i * NREL_ * DK_, FP(DC_BB) + i * H_ * NREL_ };
        run_attention(stream, wc, XD, XE, TRG_, SRC_, d2e_rel, src, 0,
                      Qb, Kb, Vb, Sb, T1, Qb);
        add_ln_kernel<<<B_ * TRG_, 128, 0, stream>>>(XD, Qb, FP(DL2_G) + i * D_, FP(DL2_B) + i * D_);

        run_ffn(stream, XD, B_ * TRG_, FP(DF_W1) + i * DF, FP(DF_B1) + i * F_,
                FP(DF_W2) + i * DF, FP(DF_B2) + i * D_, T0, T1);
        add_ln_kernel<<<B_ * TRG_, 128, 0, stream>>>(XD, T1, FP(DL3_G) + i * D_, FP(DL3_B) + i * D_);
    }

    // ---------------- Final projection to vocab ----------------
    launch_gemm(stream, XD, FP(FC_W), FP(FC_B), (float*)d_out,
                B_ * TRG_, V_, D_, D_, V_, V_, 0,0,0,0,0,0, 1,1,0,0);
}
